// RelativeGlobalAttention_49821620633749
// MI455X (gfx1250) — matrix operand order repaired and checked by reading, not compiled, not run on hardware
//
#include <hip/hip_runtime.h>

namespace {
constexpr int kDModel = 1024;
constexpr int kHeads  = 16;
constexpr int kDHead  = 64;
constexpr int kMaxLen = 16384;
constexpr int kBatch  = 4;
constexpr int kSeq    = 1024;
constexpr int kRows   = 32;   // score rows handled per attention block
constexpr int kQRows  = 48;   // Q rows staged: need [s0, s0+kRows] -> 33, padded to 3 tiles
}

typedef _Float16 half8  __attribute__((ext_vector_type(8)));
typedef _Float16 half16 __attribute__((ext_vector_type(16)));
typedef float    v8f    __attribute__((ext_vector_type(8)));
typedef int      v4i    __attribute__((ext_vector_type(4)));

#if __has_builtin(__builtin_amdgcn_global_load_async_to_lds_b128) && \
    __has_builtin(__builtin_amdgcn_s_wait_asynccnt)
#define USE_ASYNC_LDS 1
#else
#define USE_ASYNC_LDS 0
#endif

// 16-byte global -> LDS copy; async (ASYNCcnt) when the toolchain exposes it.
__device__ __forceinline__ void copy16(void* lds, const void* g) {
#if USE_ASYNC_LDS
  __builtin_amdgcn_global_load_async_to_lds_b128(
      (__attribute__((address_space(1))) v4i*)g,
      (__attribute__((address_space(3))) v4i*)lds, 0, 0);
#else
  *(uint4*)lds = *(const uint4*)g;
#endif
}

__device__ __forceinline__ void wait_async() {
#if USE_ASYNC_LDS
  __builtin_amdgcn_s_wait_asynccnt(0);
#endif
}

__device__ __forceinline__ v8f wmma_f32_16x16x32(half16 a, half16 b, v8f c) {
  // (neg_a, A, neg_b, B, c_mod, C, reuse_a, reuse_b)
  return __builtin_amdgcn_wmma_f32_16x16x32_f16(false, a, false, b, (short)0, c,
                                                false, false);
}

// A-fragment (16x32 f16) from an LDS tile, row stride `stride` halves, K-chunk base kb.
// ISA layout: lane m=lane&15; halves 0..7 = K koff+0..7, halves 8..15 = K koff+16..23,
// koff = 8*(lane>=16).
__device__ __forceinline__ half16 frag_a(const _Float16* tile, int stride, int kb) {
  int lane = threadIdx.x & 31;
  int m = lane & 15;
  int koff = (lane < 16) ? 0 : 8;
  const _Float16* p = tile + m * stride + kb + koff;
  half8 lo = *(const half8*)(p);
  half8 hi = *(const half8*)(p + 16);
  half16 a;
#pragma unroll
  for (int i = 0; i < 8; ++i) { a[i] = lo[i]; a[8 + i] = hi[i]; }
  return a;
}

// Same A-fragment but sourced from f32 LDS (softmax probabilities), converting to f16.
__device__ __forceinline__ half16 frag_a_f32(const float* tile, int stride, int kb) {
  int lane = threadIdx.x & 31;
  int m = lane & 15;
  int koff = (lane < 16) ? 0 : 8;
  const float* p = tile + m * stride + kb + koff;
  half16 a;
#pragma unroll
  for (int i = 0; i < 8; ++i) {
    a[i]     = (_Float16)p[i];
    a[8 + i] = (_Float16)p[16 + i];
  }
  return a;
}

// B-fragment (32x16 f16): lane holds column n=lane&15; halves 0..7 = K koff+0..7,
// halves 8..15 = K koff+16..23, koff = 8*(lane>=16).
// Tile must be stored [n][k] (column-of-output major), row stride `stride` halves.
__device__ __forceinline__ half16 frag_b(const _Float16* tile, int stride, int kb) {
  int lane = threadIdx.x & 31;
  int n = lane & 15;
  int k0 = kb + ((lane < 16) ? 0 : 8);
  const _Float16* p = tile + n * stride + k0;
  half8 lo = *(const half8*)(p);
  half8 hi = *(const half8*)(p + 16);
  half16 b;
#pragma unroll
  for (int i = 0; i < 8; ++i) { b[i] = lo[i]; b[8 + i] = hi[i]; }
  return b;
}

// ---------------- conversion kernels ----------------

__global__ void k_f32_to_f16(const float* __restrict__ src, _Float16* __restrict__ dst,
                             int n) {
  for (int i = blockIdx.x * blockDim.x + threadIdx.x; i < n;
       i += gridDim.x * blockDim.x)
    dst[i] = (_Float16)src[i];
}

// WT[n][k] = W[k][n]  (1024x1024) so the GEMM B-fragment reads contiguous K per column.
__global__ void k_transpose_f16(const float* __restrict__ W, _Float16* __restrict__ WT) {
  for (int i = blockIdx.x * blockDim.x + threadIdx.x; i < kDModel * kDModel;
       i += gridDim.x * blockDim.x) {
    int n = i >> 10, k = i & (kDModel - 1);
    WT[i] = (_Float16)W[k * kDModel + n];
  }
}

__global__ void k_er_f16(const float* __restrict__ Er, _Float16* __restrict__ ErH) {
  for (int i = blockIdx.x * blockDim.x + threadIdx.x; i < kSeq * kDHead;
       i += gridDim.x * blockDim.x)
    ErH[i] = (_Float16)Er[(size_t)(kMaxLen - kSeq) * kDHead + i];
}

// ---------------- projection GEMM ----------------
// C[4096 x 1024] = X @ W + bias, X f16, WT f16 ([n][k]).
// Block tile 64 rows x 128 cols, 8 waves; double-buffered LDS with async-to-LDS staging.
// mode 0: out [B,H,S,Dh]; mode 1: out [B,H,Dh,S].
__global__ __launch_bounds__(256) void k_proj(const _Float16* __restrict__ X,
                                              const _Float16* __restrict__ WT,
                                              const float* __restrict__ bias,
                                              _Float16* __restrict__ out, int mode) {
  __shared__ __attribute__((aligned(32))) _Float16 As[2][64 * 32];
  __shared__ __attribute__((aligned(32))) _Float16 Bs[2][128 * 32];

  const int tid  = threadIdx.x;
  const int wave = tid >> 5;
  const int lane = tid & 31;
  const int rb   = blockIdx.x >> 3;   // 64 row-blocks of 64
  const int cb   = blockIdx.x & 7;    // 8 col-blocks of 128
  const int row0 = rb * 64, col0 = cb * 128;
  const int rt = wave & 3;            // row tile within block (16 rows)
  const int cs = wave >> 2;           // 64-col strip

  auto stage = [&](int k0, int buf) {
    {  // A tile: 64 rows x 32 halves = 256 x b128, one per thread
      int r = tid >> 2, cc = tid & 3;
      copy16(&As[buf][r * 32 + cc * 8],
             &X[(size_t)(row0 + r) * kDModel + k0 + cc * 8]);
    }
#pragma unroll
    for (int i = 0; i < 2; ++i) {  // B tile: 128 rows x 32 halves = 512 x b128
      int idx = tid + i * 256;
      int r = idx >> 2, cc = idx & 3;
      copy16(&Bs[buf][r * 32 + cc * 8],
             &WT[(size_t)(col0 + r) * kDModel + k0 + cc * 8]);
    }
  };

  v8f acc[4] = {};

  stage(0, 0);
  wait_async();
  __syncthreads();

  for (int k0 = 0; k0 < kDModel; k0 += 32) {
    const int buf = (k0 >> 5) & 1;
    if (k0 + 32 < kDModel) stage(k0 + 32, buf ^ 1);

    half16 a = frag_a(As[buf] + rt * 16 * 32, 32, 0);
#pragma unroll
    for (int ct = 0; ct < 4; ++ct) {
      half16 b = frag_b(Bs[buf] + (cs * 64 + ct * 16) * 32, 32, 0);
      acc[ct] = wmma_f32_16x16x32(a, b, acc[ct]);
    }
    wait_async();
    __syncthreads();
  }

  const int n = lane & 15;
  const int hiOff = (lane >> 4) * 8;
#pragma unroll
  for (int ct = 0; ct < 4; ++ct) {
    int gcol = col0 + cs * 64 + ct * 16 + n;
    float bv = bias[gcol];
    int h = gcol >> 6, dh = gcol & 63;
#pragma unroll
    for (int r = 0; r < 8; ++r) {
      int m = r + hiOff;                       // D layout: M = r + 8*(lane>=16)
      int g = row0 + rt * 16 + m;
      int b = g >> 10, s = g & (kSeq - 1);
      float val = acc[ct][r] + bv;
      size_t o;
      if (mode == 0)
        o = ((size_t)((b * kHeads + h) * kSeq + s)) * kDHead + dh;   // [B,H,S,Dh]
      else
        o = ((size_t)((b * kHeads + h) * kDHead + dh)) * kSeq + s;   // [B,H,Dh,S]
      out[o] = (_Float16)val;
    }
  }
}

// ---------------- fused relative attention ----------------
// One block per (b,h, 32-row slab). Dynamic LDS:
//   sc[32*1024] f32 (128KB) | qa[48*64] f16 (6KB) | kv[2][64*64] f16 (16KB)
__global__ __launch_bounds__(256) void k_attn(const _Float16* __restrict__ Qh,
                                              const _Float16* __restrict__ Kh,
                                              const _Float16* __restrict__ VhT,
                                              const _Float16* __restrict__ ErH,
                                              float* __restrict__ out) {
  extern __shared__ __attribute__((aligned(32))) char smem[];
  float*    sc = (float*)smem;                                   // kRows x kSeq
  _Float16* qa = (_Float16*)(smem + (size_t)kRows * kSeq * 4);   // kQRows x 64
  _Float16* kv0 = qa + kQRows * 64;                              // 64 x 64, buffer 0
  _Float16* kv1 = kv0 + 64 * 64;                                 // 64 x 64, buffer 1

  const int tid  = threadIdx.x;
  const int wave = tid >> 5;
  const int lane = tid & 31;
  const int bh = blockIdx.x >> 5;                 // 32 slabs per (b,h)
  const int s0 = (blockIdx.x & 31) * kRows;
  const int b = bh >> 4, h = bh & 15;

  // Stage a 64x64 f16 tile (rows contiguous by 64 halves) into kv buffer `buf`.
  auto stage_tile = [&](const _Float16* gbase, size_t rowStride, int buf) {
    _Float16* dst = buf ? kv1 : kv0;
    for (int idx = tid; idx < 64 * 8; idx += 256) {
      int r = idx >> 3, cc = idx & 7;
      copy16(&dst[r * 64 + cc * 8], gbase + (size_t)r * rowStride + cc * 8);
    }
  };

  // Stage Q rows [s0, s0+48) (zero-pad beyond kSeq; rows 33..47 only feed dead lanes).
  for (int idx = tid; idx < kQRows * 8; idx += 256) {
    int r = idx >> 3, cc = idx & 7;
    int s = s0 + r;
    uint4 v = make_uint4(0u, 0u, 0u, 0u);
    if (s < kSeq) v = *(const uint4*)(&Qh[((size_t)bh * kSeq + s) * kDHead + cc * 8]);
    *(uint4*)(&qa[r * 64 + cc * 8]) = v;
  }

  // ---- Phase 1: scores = Q K^T ----
  {
    const int rt = wave >> 2, ct = wave & 3;      // 2 x 4 tile grid per 64-col chunk
    const _Float16* kBase = Kh + (size_t)bh * kSeq * kDHead;
    stage_tile(kBase, kDHead, 0);
    wait_async();
    __syncthreads();
    for (int tc = 0; tc < kSeq / 64; ++tc) {
      const int buf = tc & 1;
      if (tc + 1 < kSeq / 64)
        stage_tile(kBase + (size_t)(tc + 1) * 64 * kDHead, kDHead, buf ^ 1);
      const _Float16* kvb = buf ? kv1 : kv0;
      v8f c = {};
      c = wmma_f32_16x16x32(frag_a(qa + rt * 16 * 64, 64, 0),
                            frag_b(kvb + ct * 16 * 64, 64, 0), c);
      c = wmma_f32_16x16x32(frag_a(qa + rt * 16 * 64, 64, 32),
                            frag_b(kvb + ct * 16 * 64, 64, 32), c);
      const int n = lane & 15, hiOff = (lane >> 4) * 8;
#pragma unroll
      for (int r = 0; r < 8; ++r)
        sc[(rt * 16 + r + hiOff) * kSeq + tc * 64 + ct * 16 + n] = c[r];
      wait_async();
      __syncthreads();
    }
  }

  // ---- Phase 2: scatter-add the skewed Q·Er^T (Srel) ----
  // Bijection: QEr[r,j] -> (row=r,   t=j-(S-1-r)) if j >= S-1-r
  //                        (row=r-1, t=j+r+1)     otherwise;  Srel[s,s+1] = 0.
  {
    stage_tile(ErH, kDHead, 0);
    wait_async();
    __syncthreads();
    for (int jc = 0; jc < kSeq / 64; ++jc) {
      const int buf = jc & 1;
      if (jc + 1 < kSeq / 64)
        stage_tile(ErH + (size_t)(jc + 1) * 64 * kDHead, kDHead, buf ^ 1);
      const _Float16* kvb = buf ? kv1 : kv0;
      for (int t8 = wave; t8 < 12; t8 += 8) {     // 3 row-tiles x 4 col-tiles
        const int rt = t8 >> 2, ct = t8 & 3;
        v8f c = {};
        c = wmma_f32_16x16x32(frag_a(qa + rt * 16 * 64, 64, 0),
                              frag_b(kvb + ct * 16 * 64, 64, 0), c);
        c = wmma_f32_16x16x32(frag_a(qa + rt * 16 * 64, 64, 32),
                              frag_b(kvb + ct * 16 * 64, 64, 32), c);
        const int n = lane & 15, hiOff = (lane >> 4) * 8;
        const int j = jc * 64 + ct * 16 + n;
#pragma unroll
        for (int r = 0; r < 8; ++r) {
          int mrel = rt * 16 + r + hiOff;         // 0..47
          int rr = s0 + mrel;
          if (rr >= kSeq) continue;
          int row, t;
          if (j >= kSeq - 1 - rr) { row = mrel;     t = j - (kSeq - 1 - rr); }
          else                    { row = mrel - 1; t = j + rr + 1; }
          if (row >= 0 && row < kRows) sc[row * kSeq + t] += c[r];  // unique writer
        }
      }
      wait_async();
      __syncthreads();
    }
  }

  // ---- Phase 3: softmax over each row, scale 1/sqrt(Dh)=1/8 folded into exp2 ----
  {
    const float kScale = 0.125f * 1.4426950408889634f;  // log2(e)/8
#pragma unroll
    for (int rl = 0; rl < kRows / 8; ++rl) {
      int row = rl * 8 + wave;
      float* rp = sc + row * kSeq;
      float mx = -3.0e38f;
      for (int i = lane; i < kSeq; i += 32) mx = fmaxf(mx, rp[i]);
      for (int off = 16; off; off >>= 1) mx = fmaxf(mx, __shfl_xor(mx, off, 32));
      float sum = 0.f;
      for (int i = lane; i < kSeq; i += 32) {
        float p = exp2f((rp[i] - mx) * kScale);
        rp[i] = p;
        sum += p;
      }
      for (int off = 16; off; off >>= 1) sum += __shfl_xor(sum, off, 32);
      float inv = 1.0f / sum;
      for (int i = lane; i < kSeq; i += 32) rp[i] *= inv;
    }
  }
  __syncthreads();

  // ---- Phase 4: out = P @ V  (V staged as [d][t], B-fragment-native) ----
  {
    const int rt = wave >> 2, ct = wave & 3;
    const _Float16* vBase = VhT + (size_t)bh * kDHead * kSeq;
    v8f acc = {};
    stage_tile(vBase, kSeq, 0);      // rows d, row stride kSeq, cols t in [0,64)
    wait_async();
    __syncthreads();
    for (int tc = 0; tc < kSeq / 64; ++tc) {
      const int buf = tc & 1;
      if (tc + 1 < kSeq / 64)
        stage_tile(vBase + (size_t)(tc + 1) * 64, kSeq, buf ^ 1);
      const _Float16* kvb = buf ? kv1 : kv0;
      acc = wmma_f32_16x16x32(frag_a_f32(sc + rt * 16 * kSeq, kSeq, tc * 64),
                              frag_b(kvb + ct * 16 * 64, 64, 0), acc);
      acc = wmma_f32_16x16x32(frag_a_f32(sc + rt * 16 * kSeq, kSeq, tc * 64 + 32),
                              frag_b(kvb + ct * 16 * 64, 64, 32), acc);
      wait_async();
      __syncthreads();
    }
    const int n = lane & 15, hiOff = (lane >> 4) * 8;
#pragma unroll
    for (int r = 0; r < 8; ++r) {
      int s = s0 + rt * 16 + r + hiOff;
      int col = h * kDHead + ct * 16 + n;
      out[((size_t)b * kSeq + s) * kDModel + col] = acc[r];
    }
  }
}

// ---------------- host side ----------------

extern "C" void kernel_launch(void* const* d_in, const int* in_sizes, int n_in,
                              void* d_out, int out_size, void* d_ws, size_t ws_size,
                              hipStream_t stream) {
  (void)in_sizes; (void)n_in; (void)out_size; (void)ws_size;

  const float* q_in = (const float*)d_in[0];
  const float* k_in = (const float*)d_in[1];
  const float* v_in = (const float*)d_in[2];
  const float* Wq   = (const float*)d_in[3];
  const float* bq   = (const float*)d_in[4];
  const float* Wk   = (const float*)d_in[5];
  const float* bk   = (const float*)d_in[6];
  const float* Wv   = (const float*)d_in[7];
  const float* bv   = (const float*)d_in[8];
  const float* Er   = (const float*)d_in[9];
  float* out = (float*)d_out;

  char* ws = (char*)d_ws;
  size_t off = 0;
  auto carve = [&](size_t bytes) {
    char* p = ws + off;
    off += (bytes + 255) & ~(size_t)255;
    return p;
  };
  const size_t nTok = (size_t)kBatch * kSeq;          // 4096
  _Float16* Xq  = (_Float16*)carve(nTok * kDModel * 2);
  _Float16* Xk  = (_Float16*)carve(nTok * kDModel * 2);
  _Float16* Xv  = (_Float16*)carve(nTok * kDModel * 2);
  _Float16* WqT = (_Float16*)carve((size_t)kDModel * kDModel * 2);
  _Float16* WkT = (_Float16*)carve((size_t)kDModel * kDModel * 2);
  _Float16* WvT = (_Float16*)carve((size_t)kDModel * kDModel * 2);
  _Float16* ErH = (_Float16*)carve((size_t)kSeq * kDHead * 2);
  _Float16* Qh  = (_Float16*)carve(nTok * kDModel * 2);   // [B,H,S,Dh]
  _Float16* Kh  = (_Float16*)carve(nTok * kDModel * 2);   // [B,H,S,Dh]
  _Float16* VhT = (_Float16*)carve(nTok * kDModel * 2);   // [B,H,Dh,S]

  const int nElem = (int)(nTok * kDModel);
  k_f32_to_f16<<<1024, 256, 0, stream>>>(q_in, Xq, nElem);
  k_f32_to_f16<<<1024, 256, 0, stream>>>(k_in, Xk, nElem);
  k_f32_to_f16<<<1024, 256, 0, stream>>>(v_in, Xv, nElem);
  k_transpose_f16<<<1024, 256, 0, stream>>>(Wq, WqT);
  k_transpose_f16<<<1024, 256, 0, stream>>>(Wk, WkT);
  k_transpose_f16<<<1024, 256, 0, stream>>>(Wv, WvT);
  k_er_f16<<<256, 256, 0, stream>>>(Er, ErH);

  k_proj<<<512, 256, 0, stream>>>(Xq, WqT, bq, Qh, 0);
  k_proj<<<512, 256, 0, stream>>>(Xk, WkT, bk, Kh, 0);
  k_proj<<<512, 256, 0, stream>>>(Xv, WvT, bv, VhT, 1);

  const size_t smem = (size_t)kRows * kSeq * 4 + (size_t)kQRows * 64 * 2 +
                      2 * (size_t)64 * 64 * 2;
  k_attn<<<kBatch * kHeads * (kSeq / kRows), 256, smem, stream>>>(Qh, Kh, VhT, ErH, out);
}
